// DynamicSemanticChunker_111669150374
// MI455X (gfx1250) — matrix operand order repaired and checked by reading, not compiled, not run on hardware
//
#include <hip/hip_runtime.h>
#include <hip/hip_bf16.h>
#include <math.h>
#include <stdint.h>

typedef __attribute__((ext_vector_type(16))) __bf16 v16bf;
typedef __attribute__((ext_vector_type(4)))  __bf16 v4bf;
typedef __attribute__((ext_vector_type(8)))  float  v8f;

#define B_DIM 8
#define S_DIM 2048
#define D_DIM 1024
#define H_DIM 512
#define M_DIM (B_DIM * S_DIM)   // 16384 tokens

#define TM 64
#define TN 128
#define TK 32
#define LDSS (TK + 8)           // padded LDS row stride (bf16 elems); 80B, multiple of 16B

__device__ __forceinline__ void splitf(float x, __bf16 &h, __bf16 &l) {
    h = (__bf16)x;
    l = (__bf16)(x - (float)h);
}

// CDNA5 async copy: global -> LDS, 16 bytes per lane, tracked by ASYNCcnt (in-order).
__device__ __forceinline__ void async_copy_b128(uint32_t lds_off, const void* gptr) {
    asm volatile("global_load_async_to_lds_b128 %0, %1, off"
                 :: "v"(lds_off), "v"(gptr) : "memory");
}
__device__ __forceinline__ void wait_async0() {
    asm volatile("s_wait_asynccnt 0x0" ::: "memory");
}
// Oldest batch (6 copies/tile) retired; newest 6 may remain in flight.
__device__ __forceinline__ void wait_async6() {
    asm volatile("s_wait_asynccnt 0x6" ::: "memory");
}

// One-shot fp32 -> (hi, lo) bf16 split of a whole tensor (4 floats/thread).
__global__ __launch_bounds__(256)
void split_kernel(const float* __restrict__ in, __bf16* __restrict__ hi,
                  __bf16* __restrict__ lo, int n4) {
    const int i = blockIdx.x * blockDim.x + threadIdx.x;
    if (i >= n4) return;
    float4 v = ((const float4*)in)[i];
    float vs[4] = {v.x, v.y, v.z, v.w};
    v4bf h, l;
    #pragma unroll
    for (int e = 0; e < 4; ++e) {
        __bf16 hh, ll;
        splitf(vs[e], hh, ll);
        h[e] = hh;
        l[e] = ll;
    }
    ((v4bf*)hi)[i] = h;
    ((v4bf*)lo)[i] = l;
}

// C[M x N] = act(X @ W^T + bias) with pre-split hi/lo bf16 operands.
// Tiles staged via global_load_async_to_lds_b128, triple-buffered (2-deep prefetch).
__global__ __launch_bounds__(256)
void gemm_bf16_async(const __bf16* __restrict__ Xhi, const __bf16* __restrict__ Xlo,
                     const __bf16* __restrict__ Whi, const __bf16* __restrict__ Wlo,
                     const float* __restrict__ bias, float* __restrict__ C,
                     int N, int relu) {
    __shared__ __align__(16) __bf16 As[3][2][TM][LDSS];   // [buf][hi/lo][row][k]
    __shared__ __align__(16) __bf16 Bs[3][2][TN][LDSS];

    const int tid  = threadIdx.x;
    const int lane = tid & 31;
    const int wid  = tid >> 5;
    const int wm   = wid >> 2;              // 0..1  (M waves)
    const int wn   = wid & 3;               // 0..3  (N waves)
    const int row0 = blockIdx.y * TM;
    const int col0 = blockIdx.x * TN;

    const v8f vzero = {0.f,0.f,0.f,0.f,0.f,0.f,0.f,0.f};
    v8f acc[2][2];
    #pragma unroll
    for (int i = 0; i < 2; ++i)
        #pragma unroll
        for (int j = 0; j < 2; ++j) acc[i][j] = vzero;

    // Per k-step: 6 async b128 issues/thread (A hi+lo: 1 each, B hi+lo: 2 each).
    auto issue_async = [&](int buf, int kt) {
        const int arow = tid >> 2;                 // 0..63
        const int ak   = (tid & 3) * 8;            // 0,8,16,24
        const size_t ga = (size_t)(row0 + arow) * D_DIM + (size_t)kt * TK + ak;
        async_copy_b128((uint32_t)(uintptr_t)&As[buf][0][arow][ak], Xhi + ga);
        async_copy_b128((uint32_t)(uintptr_t)&As[buf][1][arow][ak], Xlo + ga);
        const int brow = tid >> 1;                 // 0..127
        const int bk0  = (tid & 1) * 16;           // 0,16
        const size_t gb = (size_t)(col0 + brow) * D_DIM + (size_t)kt * TK + bk0;
        async_copy_b128((uint32_t)(uintptr_t)&Bs[buf][0][brow][bk0],     Whi + gb);
        async_copy_b128((uint32_t)(uintptr_t)&Bs[buf][0][brow][bk0 + 8], Whi + gb + 8);
        async_copy_b128((uint32_t)(uintptr_t)&Bs[buf][1][brow][bk0],     Wlo + gb);
        async_copy_b128((uint32_t)(uintptr_t)&Bs[buf][1][brow][bk0 + 8], Wlo + gb + 8);
    };

    // ISA 7.12.2: 16-bit A 16x32 lane layout
    auto load_a = [&](int buf, int hl, int mt) -> v16bf {
        v16bf a;
        const int arow = wm * 32 + mt * 16 + (lane & 15);
        const int kadd = (lane >> 4) * 8;
        #pragma unroll
        for (int e = 0; e < 16; ++e) {
            const int j = e >> 1, p = e & 1;
            const int kidx = ((j < 4) ? (2 * j + p) : (16 + 2 * (j - 4) + p)) + kadd;
            a[e] = As[buf][hl][arow][kidx];
        }
        return a;
    };
    // ISA 7.12.2: 16-bit B 32x16 lane layout (cols across lanes, K within lane)
    auto load_b = [&](int buf, int hl, int nt) -> v16bf {
        v16bf b;
        const int bcol  = wn * 32 + nt * 16 + (lane & 15);
        const int kbase = (lane >> 4) * 8;
        #pragma unroll
        for (int e = 0; e < 16; ++e) b[e] = Bs[buf][hl][bcol][kbase + e + ((e >= 8) ? 8 : 0)];
        return b;
    };

    const int NK = D_DIM / TK;  // 32

    issue_async(0, 0);
    issue_async(1, 1);
    wait_async6();              // tile 0 resident; tile 1 still in flight
    __syncthreads();

    int cur = 0, nxt = 2;
    for (int kt = 0; kt < NK; ++kt) {
        if (kt + 2 < NK) issue_async(nxt, kt + 2);

        v16bf ah[2], al[2], bh[2], bl[2];
        #pragma unroll
        for (int t = 0; t < 2; ++t) {
            ah[t] = load_a(cur, 0, t);
            al[t] = load_a(cur, 1, t);
            bh[t] = load_b(cur, 0, t);
            bl[t] = load_b(cur, 1, t);
        }

        #pragma unroll
        for (int mt = 0; mt < 2; ++mt)
            #pragma unroll
            for (int nt = 0; nt < 2; ++nt) {
                acc[mt][nt] = __builtin_amdgcn_wmma_f32_16x16x32_bf16(
                    false, ah[mt], false, bh[nt], (short)0, acc[mt][nt], false, false);
                acc[mt][nt] = __builtin_amdgcn_wmma_f32_16x16x32_bf16(
                    false, ah[mt], false, bl[nt], (short)0, acc[mt][nt], false, false);
                acc[mt][nt] = __builtin_amdgcn_wmma_f32_16x16x32_bf16(
                    false, al[mt], false, bh[nt], (short)0, acc[mt][nt], false, false);
            }

        // ASYNCcnt retires in order: <=6 outstanding means tile kt+1 is complete.
        if (kt + 2 < NK) wait_async6();
        else             wait_async0();
        __syncthreads();

        cur = (cur + 1 == 3) ? 0 : cur + 1;
        nxt = (nxt + 1 == 3) ? 0 : nxt + 1;
    }

    // Store: C/D layout — lanes 0-15: N=lane, M=r; lanes 16-31: N=lane-16, M=r+8
    float bv[2]; int coln[2];
    #pragma unroll
    for (int nt = 0; nt < 2; ++nt) {
        coln[nt] = col0 + wn * 32 + nt * 16 + (lane & 15);
        bv[nt] = bias[coln[nt]];
    }
    const int mbase = row0 + wm * 32 + ((lane >> 4) * 8);
    #pragma unroll
    for (int mt = 0; mt < 2; ++mt)
        #pragma unroll
        for (int nt = 0; nt < 2; ++nt)
            #pragma unroll
            for (int r = 0; r < 8; ++r) {
                float v = acc[mt][nt][r] + bv[nt];
                if (relu) v = fmaxf(v, 0.f);
                C[(size_t)(mbase + mt * 16 + r) * N + coln[nt]] = v;
            }
}

// refine[t] = sigmoid(dot(h[t,:512], w2) + b2); one wave per token
__global__ __launch_bounds__(256)
void refine_kernel(const float* __restrict__ h, const float* __restrict__ w2,
                   const float* __restrict__ b2, float* __restrict__ out) {
    const int t    = (blockIdx.x * blockDim.x + threadIdx.x) >> 5;
    const int lane = threadIdx.x & 31;
    const float4* hp = (const float4*)(h + (size_t)t * H_DIM);
    const float4* wp = (const float4*)w2;
    float s = 0.f;
    #pragma unroll
    for (int j = lane; j < H_DIM / 4; j += 32) {
        float4 a = hp[j], b = wp[j];
        s = fmaf(a.x, b.x, s); s = fmaf(a.y, b.y, s);
        s = fmaf(a.z, b.z, s); s = fmaf(a.w, b.w, s);
    }
    #pragma unroll
    for (int off = 16; off > 0; off >>= 1) s += __shfl_xor(s, off);
    if (lane == 0) out[t] = 1.f / (1.f + __expf(-(s + b2[0])));
}

// cosine-sim boundary blend; one wave per token
__global__ __launch_bounds__(256)
void boundary_kernel(const float* __restrict__ q, const float* __restrict__ k,
                     const float* __restrict__ refine,
                     float* __restrict__ fin_out, float* __restrict__ mask_out) {
    const int t    = (blockIdx.x * blockDim.x + threadIdx.x) >> 5;
    const int lane = threadIdx.x & 31;
    const int s    = t & (S_DIM - 1);
    float prob = 1.0f;
    if (s > 0) {
        const float4* qp = (const float4*)(q + (size_t)t * D_DIM);
        const float4* kp = (const float4*)(k + (size_t)(t - 1) * D_DIM);
        float dot = 0.f, nq = 0.f, nk = 0.f;
        #pragma unroll
        for (int j = lane; j < D_DIM / 4; j += 32) {
            float4 a = qp[j], b = kp[j];
            dot = fmaf(a.x, b.x, fmaf(a.y, b.y, fmaf(a.z, b.z, fmaf(a.w, b.w, dot))));
            nq  = fmaf(a.x, a.x, fmaf(a.y, a.y, fmaf(a.z, a.z, fmaf(a.w, a.w, nq))));
            nk  = fmaf(b.x, b.x, fmaf(b.y, b.y, fmaf(b.z, b.z, fmaf(b.w, b.w, nk))));
        }
        #pragma unroll
        for (int off = 16; off > 0; off >>= 1) {
            dot += __shfl_xor(dot, off);
            nq  += __shfl_xor(nq, off);
            nk  += __shfl_xor(nk, off);
        }
        const float sims = dot / (fmaxf(sqrtf(nq), 1e-8f) * fmaxf(sqrtf(nk), 1e-8f));
        prob = 0.5f * (1.0f - sims);
    }
    if (lane == 0) {
        const float fin = 0.7f * prob + 0.3f * refine[t];
        fin_out[t]  = fin;
        mask_out[t] = (fin > 0.5f) ? 1.0f : 0.0f;
    }
}

// per-row inclusive scan via ballot+popcount; one wave per batch row
__global__ __launch_bounds__(32)
void segment_kernel(const float* __restrict__ fin, float* __restrict__ seg) {
    const int b = blockIdx.x;
    const int lane = threadIdx.x;
    const unsigned lemask = (lane == 31) ? 0xffffffffu : ((1u << (lane + 1)) - 1u);
    int carry = 0;
    for (int c = 0; c < S_DIM; c += 32) {
        const int t = b * S_DIM + c + lane;
        const bool m = fin[t] > 0.5f;
        const unsigned bal = (unsigned)__ballot(m);
        int id = carry + __popc(bal & lemask) - 1;
        if (id < 0) id = 0;
        seg[t] = (float)id;
        carry += __popc(bal);
    }
}

extern "C" void kernel_launch(void* const* d_in, const int* in_sizes, int n_in,
                              void* d_out, int out_size, void* d_ws, size_t ws_size,
                              hipStream_t stream) {
    const float* X  = (const float*)d_in[0];
    const float* Wq = (const float*)d_in[1];
    const float* bq = (const float*)d_in[2];
    const float* Wk = (const float*)d_in[3];
    const float* bk = (const float*)d_in[4];
    const float* W1 = (const float*)d_in[5];
    const float* b1 = (const float*)d_in[6];
    const float* W2 = (const float*)d_in[7];
    const float* b2 = (const float*)d_in[8];

    // fp32 workspace
    float* qws = (float*)d_ws;                              // 16384x1024
    float* kws = qws + (size_t)M_DIM * D_DIM;               // 16384x1024
    float* hws = kws + (size_t)M_DIM * D_DIM;               // 16384x512
    float* rws = hws + (size_t)M_DIM * H_DIM;               // 16384
    // bf16 hi/lo workspace
    __bf16* Xhi  = (__bf16*)(rws + M_DIM);
    __bf16* Xlo  = Xhi  + (size_t)M_DIM * D_DIM;
    __bf16* Wqhi = Xlo  + (size_t)M_DIM * D_DIM;
    __bf16* Wqlo = Wqhi + (size_t)D_DIM * D_DIM;
    __bf16* Wkhi = Wqlo + (size_t)D_DIM * D_DIM;
    __bf16* Wklo = Wkhi + (size_t)D_DIM * D_DIM;
    __bf16* W1hi = Wklo + (size_t)D_DIM * D_DIM;
    __bf16* W1lo = W1hi + (size_t)H_DIM * D_DIM;

    float* fin  = (float*)d_out;            // [0 , 16384)
    float* mask = fin + M_DIM;              // [16384, 32768)
    float* seg  = mask + M_DIM;             // [32768, 49152)

    dim3 blk(256);

    const int nX  = (M_DIM * D_DIM) / 4, nWqk = (D_DIM * D_DIM) / 4, nW1 = (H_DIM * D_DIM) / 4;
    split_kernel<<<nX   / 256, blk, 0, stream>>>(X,  Xhi,  Xlo,  nX);
    split_kernel<<<nWqk / 256, blk, 0, stream>>>(Wq, Wqhi, Wqlo, nWqk);
    split_kernel<<<nWqk / 256, blk, 0, stream>>>(Wk, Wkhi, Wklo, nWqk);
    split_kernel<<<nW1  / 256, blk, 0, stream>>>(W1, W1hi, W1lo, nW1);

    dim3 gq(D_DIM / TN, M_DIM / TM);        // 8 x 256
    dim3 gh(H_DIM / TN, M_DIM / TM);        // 4 x 256
    gemm_bf16_async<<<gq, blk, 0, stream>>>(Xhi, Xlo, Wqhi, Wqlo, bq, qws, D_DIM, 0);
    gemm_bf16_async<<<gq, blk, 0, stream>>>(Xhi, Xlo, Wkhi, Wklo, bk, kws, D_DIM, 0);
    gemm_bf16_async<<<gh, blk, 0, stream>>>(Xhi, Xlo, W1hi, W1lo, b1, hws, H_DIM, 1);

    refine_kernel  <<<M_DIM / 8, blk, 0, stream>>>(hws, W2, b2, rws);
    boundary_kernel<<<M_DIM / 8, blk, 0, stream>>>(qws, kws, rws, fin, mask);
    segment_kernel <<<B_DIM, 32, 0, stream>>>(fin, seg);
}
